// MultiHeadSelfAttention_42159398977859
// MI455X (gfx1250) — matrix operand order repaired and checked by reading, not compiled, not run on hardware
//
#include <hip/hip_runtime.h>

#define DEV __device__ __forceinline__

typedef __attribute__((ext_vector_type(4)))  unsigned short u16x4;
typedef __attribute__((ext_vector_type(8)))  unsigned short u16x8;
typedef __attribute__((ext_vector_type(16))) unsigned short u16x16;
typedef __attribute__((ext_vector_type(2)))  __bf16        v2bf;
typedef __attribute__((ext_vector_type(16))) __bf16        v16bf;
typedef __attribute__((ext_vector_type(8)))  float         v8f;
typedef __attribute__((ext_vector_type(2)))  unsigned int  u32x2;
typedef __attribute__((ext_vector_type(4)))  unsigned int  u32x4;

// exact pointee type from the async builtin's prototype
typedef int i32x4 __attribute__((vector_size(16)));
typedef __attribute__((address_space(1))) i32x4* g_i32x4_p;   // global
typedef __attribute__((address_space(3))) i32x4* l_i32x4_p;   // LDS

#if defined(__has_builtin)
#if __has_builtin(__builtin_amdgcn_global_load_async_to_lds_b128)
#define ASYNC_LDS 1
#endif
#if __has_builtin(__builtin_amdgcn_tensor_load_to_lds)
#define HAVE_TDM 1
#endif
#endif

// 16-byte global -> LDS copy: async (ASYNCcnt-tracked, no VGPR round trip).
DEV void copy_b128(const unsigned short* g, unsigned short* l) {
#ifdef ASYNC_LDS
    __builtin_amdgcn_global_load_async_to_lds_b128(
        (g_i32x4_p)g, (l_i32x4_p)l, 0, 0);
#else
    *(u16x8*)l = *(const u16x8*)g;
#endif
}
DEV void copy_b256(const unsigned short* g, unsigned short* l) {
    copy_b128(g, l);
    copy_b128(g + 8, l + 8);
}
DEV void wait_async() {
#ifdef ASYNC_LDS
#if __has_builtin(__builtin_amdgcn_s_wait_asynccnt)
    __builtin_amdgcn_s_wait_asynccnt(0);
#else
    asm volatile("s_wait_asynccnt 0x0" ::: "memory");
#endif
#endif
}

#ifdef HAVE_TDM
typedef __attribute__((ext_vector_type(4))) unsigned int tdmv4u;
typedef __attribute__((ext_vector_type(8))) int          tdmv8i;
typedef __attribute__((ext_vector_type(4))) int          tdmv4i;

DEV unsigned lds_addr_of(const void* p) {   // DS-relative byte address
    return (unsigned)(unsigned long long)(__attribute__((address_space(3))) const void*)p;
}

// 2D TDM tile load of 2-byte elements: tile_d0 x tile_d1, row stride = stride0
// (all in elements), packed contiguously into LDS at ldsoff. Issue once per
// workgroup (EXEC-independent); completion via TENSORcnt.
DEV void tdm_load_2d(const unsigned short* gaddr, unsigned ldsoff,
                     unsigned tile_d0, unsigned tile_d1,
                     unsigned tensor_d0, unsigned tensor_d1,
                     unsigned long long stride0) {
    unsigned long long ga = (unsigned long long)gaddr;
    tdmv4u g0;
    g0[0] = 1u;                                        // count=1, user mode
    g0[1] = ldsoff;                                    // lds_addr (bytes)
    g0[2] = (unsigned)(ga & 0xFFFFFFFFu);              // global_addr[31:0]
    g0[3] = (unsigned)((ga >> 32) & 0x01FFFFFFu)       // global_addr[56:32]
          | (2u << 30);                                // type = 2 ("image")
    tdmv8i g1;
    g1[0] = 0x10000;                                   // data_size=1 (2 bytes)
    g1[1] = (int)((tensor_d0 & 0xFFFFu) << 16);        // tensor_dim0[15:0]
    g1[2] = (int)(((tensor_d0 >> 16) & 0xFFFFu) | ((tensor_d1 & 0xFFFFu) << 16));
    g1[3] = (int)(((tensor_d1 >> 16) & 0xFFFFu) | ((tile_d0 & 0xFFFFu) << 16));
    g1[4] = (int)(tile_d1 & 0xFFFFu);                  // tile_dim1; tile_dim2=0
    g1[5] = (int)(stride0 & 0xFFFFFFFFull);            // dim0 stride [31:0]
    g1[6] = (int)((stride0 >> 32) & 0xFFFFull);        // dim0 stride [47:32]
    g1[7] = 0;
    tdmv4i z4 = {};
    tdmv8i z8 = {};
    // 6-arg form on this toolchain: (g0, g1, g2, g3, <v8i>, cpol); groups 2/3
    // unused for 2D tensors -> zero-filled.
    __builtin_amdgcn_tensor_load_to_lds(g0, g1, z4, z4, z8, 0);
}
DEV void wait_tensor() {
#if __has_builtin(__builtin_amdgcn_s_wait_tensorcnt)
    __builtin_amdgcn_s_wait_tensorcnt(0);
#else
    asm volatile("s_wait_tensorcnt 0x0" ::: "memory");
#endif
}
#endif

// f32 -> bf16 via native fptrunc (lets the backend use HW cvt if present)
DEV unsigned short f32_to_bf16(float f) {
    __bf16 h = (__bf16)f;
    return __builtin_bit_cast(unsigned short, h);
}
DEV unsigned pk_bf16(float a, float b) {
    v2bf t; t[0] = (__bf16)a; t[1] = (__bf16)b;
    return __builtin_bit_cast(unsigned, t);
}

DEV float fast_exp2(float x) {
#if defined(__has_builtin)
#if __has_builtin(__builtin_amdgcn_exp2f)
    return __builtin_amdgcn_exp2f(x);    // v_exp_f32 is natively exp2
#else
    return exp2f(x);
#endif
#else
    return exp2f(x);
#endif
}

// A-operand (16x32 bf16): lane (half,r) holds row r, K = half*8+{0..7} and 16+half*8+{0..7}
DEV v16bf load_a_frag(const unsigned short* row_base, int half) {
    u16x8 lo = *(const u16x8*)(row_base + half * 8);
    u16x8 hi = *(const u16x8*)(row_base + 16 + half * 8);
    u16x16 w;
#pragma unroll
    for (int i = 0; i < 8; ++i) { w[i] = lo[i]; w[i + 8] = hi[i]; }
    return __builtin_bit_cast(v16bf, w);
}

// B-operand (32x16 bf16): lane (half,r) holds column r, K = half*8+{0..7} and 16+half*8+{0..7}
DEV v16bf load_b_frag(const unsigned short* col_base, int half) {
    u16x8 lo = *(const u16x8*)(col_base + half * 8);
    u16x8 hi = *(const u16x8*)(col_base + 16 + half * 8);
    u16x16 w;
#pragma unroll
    for (int i = 0; i < 8; ++i) { w[i] = lo[i]; w[i + 8] = hi[i]; }
    return __builtin_bit_cast(v16bf, w);
}

// ---------------------------------------------------------------- cast kernels
__global__ void cast_f32_bf16(const float* __restrict__ in,
                              unsigned short* __restrict__ out, int n) {
    int i = (blockIdx.x * blockDim.x + threadIdx.x) * 4;
    if (i < n) {
        float4 f = *(const float4*)(in + i);
        u32x2 o = { pk_bf16(f.x, f.y), pk_bf16(f.z, f.w) };
        *(u32x2*)(out + i) = o;
    }
}

// w [K,N] f32 row-major -> wT [N,K] bf16 (GEMM B tiles become K-contiguous)
__global__ void castT_f32_bf16(const float* __restrict__ in,
                               unsigned short* __restrict__ out, int K, int N) {
    int idx = blockIdx.x * blockDim.x + threadIdx.x;
    if (idx < K * N) {
        int k = idx / N, n = idx - k * N;
        out[(size_t)n * K + k] = f32_to_bf16(in[idx]);
    }
}

// ---------------------------------------------------------------- GEMM
// C[M,N] = A[M,K](bf16) * B[K,N] + bias[N];  B pre-transposed as Bt[N,K].
// Columns < scale_cols get multiplied by `scale` after bias (folds the attention
// score scale 0.125*log2(e) into the Q columns of the QKV projection).
// Block tile 128x128, 8 waves (4Mx2N), wave tile 32x64 (2x4 WMMA tiles).
// Double-buffered LDS; tile staging by TDM (one tensor_load_to_lds per tile,
// issued by wave 0) or per-thread async-to-LDS fallback; ONE barrier per K-step.
template <bool OUT_BF16>
__global__ __launch_bounds__(256) void gemm_bf16(
    const unsigned short* __restrict__ A,
    const unsigned short* __restrict__ Bt,
    const float* __restrict__ bias,
    void* __restrict__ outp, int M, int N, int K,
    int scale_cols, float scale)
{
    __shared__ unsigned short lgA[2][128 * 32];   // [buf][m][k]
    __shared__ unsigned short lgB[2][128 * 32];   // [buf][n][k]
    const int tid  = threadIdx.x;
    const int lane = tid & 31, wave = tid >> 5;
    const int half = lane >> 4, r = lane & 15;
    const int wm = wave & 3, wn = wave >> 2;          // 4 x 2 wave grid
    const int m0 = blockIdx.y * 128, n0 = blockIdx.x * 128;

    v8f acc[2][4] = {};

#ifdef HAVE_TDM
    const unsigned short* aBase = A  + (size_t)m0 * K;
    const unsigned short* bBase = Bt + (size_t)n0 * K;
    unsigned ldsA[2] = { lds_addr_of(&lgA[0][0]), lds_addr_of(&lgA[1][0]) };
    unsigned ldsB[2] = { lds_addr_of(&lgB[0][0]), lds_addr_of(&lgB[1][0]) };
    if (wave == 0) {                                   // prologue: stage buffer 0
        tdm_load_2d(aBase, ldsA[0], 32, 128, (unsigned)K, (unsigned)M, (unsigned long long)K);
        tdm_load_2d(bBase, ldsB[0], 32, 128, (unsigned)K, (unsigned)N, (unsigned long long)K);
        wait_tensor();
    }
    __syncthreads();
#else
    const int srow = tid >> 1, sseg = (tid & 1) * 16; // 128 rows x 2 segs of 16
    const unsigned short* aSrc = A  + (size_t)(m0 + srow) * K + sseg;
    const unsigned short* bSrc = Bt + (size_t)(n0 + srow) * K + sseg;
    const int sOff = srow * 32 + sseg;
    copy_b256(aSrc, lgA[0] + sOff);                    // prologue: stage buffer 0
    copy_b256(bSrc, lgB[0] + sOff);
    wait_async();
    __syncthreads();
#endif

    for (int kb = 0; kb < K; kb += 32) {
        const int cur = (kb >> 5) & 1;
        if (kb + 32 < K) {                             // stage next tile into other buffer
#ifdef HAVE_TDM
            if (wave == 0) {
                tdm_load_2d(aBase + kb + 32, ldsA[cur ^ 1], 32, 128,
                            (unsigned)K, (unsigned)M, (unsigned long long)K);
                tdm_load_2d(bBase + kb + 32, ldsB[cur ^ 1], 32, 128,
                            (unsigned)K, (unsigned)N, (unsigned long long)K);
            }
#else
            copy_b256(aSrc + kb + 32, lgA[cur ^ 1] + sOff);
            copy_b256(bSrc + kb + 32, lgB[cur ^ 1] + sOff);
#endif
        }

        v16bf af[2], bfr[4];
#pragma unroll
        for (int mi = 0; mi < 2; ++mi)
            af[mi] = load_a_frag(lgA[cur] + (wm * 32 + mi * 16 + r) * 32, half);
#pragma unroll
        for (int ni = 0; ni < 4; ++ni)
            bfr[ni] = load_b_frag(lgB[cur] + (wn * 64 + ni * 16 + r) * 32, half);
#pragma unroll
        for (int mi = 0; mi < 2; ++mi)
#pragma unroll
            for (int ni = 0; ni < 4; ++ni)
                acc[mi][ni] = __builtin_amdgcn_wmma_f32_16x16x32_bf16(
                    false, af[mi], false, bfr[ni], (short)0, acc[mi][ni], false, false);

#ifdef HAVE_TDM
        if (wave == 0) wait_tensor();
#else
        wait_async();
#endif
        __syncthreads();   // next buffer ready; everyone done reading current
    }

#pragma unroll
    for (int mi = 0; mi < 2; ++mi) {
#pragma unroll
        for (int ni = 0; ni < 4; ++ni) {
            const int col = n0 + wn * 64 + ni * 16 + r;
            const float bv = bias[col];
            const float sc = (col < scale_cols) ? scale : 1.0f;
            float vv[8];
#pragma unroll
            for (int i = 0; i < 8; ++i) vv[i] = (acc[mi][ni][i] + bv) * sc;
            const int row0 = m0 + wm * 32 + mi * 16 + half * 8;
            if (OUT_BF16) {
                unsigned short* o = (unsigned short*)outp;
#pragma unroll
                for (int i = 0; i < 8; i += 2) {
                    unsigned w = pk_bf16(vv[i], vv[i + 1]);
                    o[(size_t)(row0 + i) * N + col]     = (unsigned short)w;
                    o[(size_t)(row0 + i + 1) * N + col] = (unsigned short)(w >> 16);
                }
            } else {
                float* o = (float*)outp;
#pragma unroll
                for (int i = 0; i < 8; ++i) o[(size_t)(row0 + i) * N + col] = vv[i];
            }
        }
    }
}

// ---------------------------------------------------------------- flash attention (transposed)
// S^T = K @ Q^T and O^T = V^T @ P^T: each lane owns ONE query column; softmax in
// the exp2 domain (Q pre-scaled by 0.125*log2e). Double-buffered tiles, async-to-
// LDS K staging, pipelined V transpose, one barrier per key block.
__global__ __launch_bounds__(256) void attn_kernel(
    const unsigned short* __restrict__ qkv,   // [B*T, 3072] bf16 (Q|K|V)
    unsigned short* __restrict__ out)         // [B*T, 1024] bf16
{
    __shared__ unsigned short lgK[2][32 * 64];     // [buf][key][d]
    __shared__ unsigned short lgVt[2][64 * 32];    // [buf][d][key]
    __shared__ unsigned short lgP[8 * 16 * 32];    // per-wave P^T: [q][32 keys]

    const int tid = threadIdx.x, lane = tid & 31, wave = tid >> 5;
    const int half = lane >> 4, r = lane & 15;
    const int h = blockIdx.y, b = blockIdx.z;
    const int q0 = blockIdx.x * 128 + wave * 16;
    const size_t rowbase = (size_t)b * 2048;
    const int qcol = h * 64, kcol = 1024 + h * 64, vcol = 2048 + h * 64;

    v16bf qf[2];
#pragma unroll
    for (int c = 0; c < 2; ++c)
        qf[c] = load_b_frag(qkv + (rowbase + q0 + r) * 3072 + qcol + c * 32, half);

    float m = -3.0e38f, l = 0.f;
    v8f ot[4] = {};                                // O^T tiles: d rows, q cols

    const int skey = tid >> 3, sseg = tid & 7;     // 32 keys x 8 segs of 8

    {   // prologue: stage block 0
        const unsigned short* kRow = qkv + (rowbase + skey) * 3072;
        copy_b128(kRow + kcol + sseg * 8, lgK[0] + skey * 64 + sseg * 8);
        u16x8 v = *(const u16x8*)(kRow + vcol + sseg * 8);
#pragma unroll
        for (int i = 0; i < 8; ++i) lgVt[0][(sseg * 8 + i) * 32 + skey] = v[i];
    }
    wait_async();
    __syncthreads();

    for (int j = 0; j < 2048; j += 32) {
        const int cur = (j >> 5) & 1;
        const bool has_next = (j + 32 < 2048);
        u16x8 vstage;
        if (has_next) {   // issue next block's loads up front
            const unsigned short* nRow = qkv + (rowbase + j + 32 + skey) * 3072;
            copy_b128(nRow + kcol + sseg * 8, lgK[cur ^ 1] + skey * 64 + sseg * 8);
            vstage = *(const u16x8*)(nRow + vcol + sseg * 8);
        }

        const unsigned short* lgKc  = lgK[cur];
        const unsigned short* lgVtc = lgVt[cur];

        // S^T[32 keys x 16 q]: batch all 8 DS frag loads, then 4 WMMAs
        v16bf ka0 = load_a_frag(lgKc + r * 64, half);
        v16bf kz0 = load_a_frag(lgKc + (16 + r) * 64, half);
        v16bf ka1 = load_a_frag(lgKc + r * 64 + 32, half);
        v16bf kz1 = load_a_frag(lgKc + (16 + r) * 64 + 32, half);
        v8f st0 = {}, st1 = {};
        st0 = __builtin_amdgcn_wmma_f32_16x16x32_bf16(false, ka0, false, qf[0], (short)0, st0, false, false);
        st1 = __builtin_amdgcn_wmma_f32_16x16x32_bf16(false, kz0, false, qf[0], (short)0, st1, false, false);
        st0 = __builtin_amdgcn_wmma_f32_16x16x32_bf16(false, ka1, false, qf[1], (short)0, st0, false, false);
        st1 = __builtin_amdgcn_wmma_f32_16x16x32_bf16(false, kz1, false, qf[1], (short)0, st1, false, false);

        // online softmax in the exp2 domain
        float s[16];
#pragma unroll
        for (int i = 0; i < 8; ++i) { s[i] = st0[i]; s[8 + i] = st1[i]; }
        float rm = s[0];
#pragma unroll
        for (int i = 1; i < 16; ++i) rm = fmaxf(rm, s[i]);
        rm = fmaxf(rm, __shfl_xor(rm, 16, 32));    // other 16 keys live in lane^16
        const float mn   = fmaxf(m, rm);
        const float corr = fast_exp2(m - mn);
        float rs = 0.f;
#pragma unroll
        for (int i = 0; i < 16; ++i) { s[i] = fast_exp2(s[i] - mn); rs += s[i]; }
        rs += __shfl_xor(rs, 16, 32);
        l = l * corr + rs;
        m = mn;
#pragma unroll
        for (int t = 0; t < 4; ++t)
#pragma unroll
            for (int i = 0; i < 8; ++i) ot[t][i] *= corr;

        // P^T -> per-wave LDS [q][key]
        unsigned short* Pw = lgP + wave * 512;
        u32x4 pa = { pk_bf16(s[0], s[1]),   pk_bf16(s[2], s[3]),
                     pk_bf16(s[4], s[5]),   pk_bf16(s[6], s[7]) };
        u32x4 pb = { pk_bf16(s[8], s[9]),   pk_bf16(s[10], s[11]),
                     pk_bf16(s[12], s[13]), pk_bf16(s[14], s[15]) };
        *(u32x4*)(Pw + r * 32 + half * 8)      = pa;
        *(u32x4*)(Pw + r * 32 + 16 + half * 8) = pb;

        // O^T[64 d x 16 q] += V^T[d x keys] @ P^T[keys x q]
        v16bf pfrag = load_b_frag(Pw + r * 32, half);
        v16bf va0 = load_a_frag(lgVtc + (0 * 16 + r) * 32, half);
        v16bf va1 = load_a_frag(lgVtc + (1 * 16 + r) * 32, half);
        v16bf va2 = load_a_frag(lgVtc + (2 * 16 + r) * 32, half);
        v16bf va3 = load_a_frag(lgVtc + (3 * 16 + r) * 32, half);
        ot[0] = __builtin_amdgcn_wmma_f32_16x16x32_bf16(false, va0, false, pfrag, (short)0, ot[0], false, false);
        ot[1] = __builtin_amdgcn_wmma_f32_16x16x32_bf16(false, va1, false, pfrag, (short)0, ot[1], false, false);
        ot[2] = __builtin_amdgcn_wmma_f32_16x16x32_bf16(false, va2, false, pfrag, (short)0, ot[2], false, false);
        ot[3] = __builtin_amdgcn_wmma_f32_16x16x32_bf16(false, va3, false, pfrag, (short)0, ot[3], false, false);

        if (has_next) {   // scatter next V block after compute
#pragma unroll
            for (int i = 0; i < 8; ++i)
                lgVt[cur ^ 1][(sseg * 8 + i) * 32 + skey] = vstage[i];
        }
        wait_async();
        __syncthreads();
    }

    const float inv = 1.f / l;
#pragma unroll
    for (int t = 0; t < 4; ++t) {
        u32x4 ov = { pk_bf16(ot[t][0] * inv, ot[t][1] * inv),
                     pk_bf16(ot[t][2] * inv, ot[t][3] * inv),
                     pk_bf16(ot[t][4] * inv, ot[t][5] * inv),
                     pk_bf16(ot[t][6] * inv, ot[t][7] * inv) };
        *(u32x4*)(out + (rowbase + q0 + r) * 1024 + h * 64 + t * 16 + half * 8) = ov;
    }
}

// ---------------------------------------------------------------- launch
extern "C" void kernel_launch(void* const* d_in, const int* in_sizes, int n_in,
                              void* d_out, int out_size, void* d_ws, size_t ws_size,
                              hipStream_t stream) {
    const float* x     = (const float*)d_in[0];   // [2,2048,1024]
    const float* w_qkv = (const float*)d_in[1];   // [1024,3072]
    const float* b_qkv = (const float*)d_in[2];   // [3072]
    const float* w_out = (const float*)d_in[3];   // [1024,1024]
    const float* b_out = (const float*)d_in[4];   // [1024]

    const int BT = 4096, D = 1024, N3 = 3072;
    char* ws = (char*)d_ws;
    unsigned short* xb    = (unsigned short*)(ws);                              //  8 MB
    unsigned short* wqkvT = (unsigned short*)(ws + (size_t)8  * 1024 * 1024);   //  6 MB
    unsigned short* woutT = (unsigned short*)(ws + (size_t)14 * 1024 * 1024);   //  2 MB
    unsigned short* qkvb  = (unsigned short*)(ws + (size_t)16 * 1024 * 1024);   // 24 MB
    unsigned short* attnb = (unsigned short*)(ws + (size_t)40 * 1024 * 1024);   //  8 MB

    cast_f32_bf16<<<(BT * D / 4 + 255) / 256, 256, 0, stream>>>(x, xb, BT * D);
    castT_f32_bf16<<<(D * N3 + 255) / 256, 256, 0, stream>>>(w_qkv, wqkvT, D, N3);
    castT_f32_bf16<<<(D * D + 255) / 256, 256, 0, stream>>>(w_out, woutT, D, D);

    // QKV projection; Q columns pre-scaled by 0.125*log2(e) for exp2-domain softmax.
    const float qscale = 0.125f * 1.4426950408889634f;
    gemm_bf16<true><<<dim3(N3 / 128, BT / 128), 256, 0, stream>>>(
        xb, wqkvT, b_qkv, qkvb, BT, N3, D, /*scale_cols=*/D, qscale);

    // fused flash attention: grid = (q-blocks, heads, batch)
    attn_kernel<<<dim3(16, 16, 2), 256, 0, stream>>>(qkvb, attnb);

    // output projection: [4096,1024] x [1024,1024] + b_out -> f32 d_out
    gemm_bf16<false><<<dim3(D / 128, BT / 128), 256, 0, stream>>>(
        attnb, woutT, b_out, d_out, BT, D, D, /*scale_cols=*/0, 1.0f);
}
